// ChebNet_76836964925956
// MI455X (gfx1250) — matrix operand order repaired and checked by reading, not compiled, not run on hardware
//
#include <hip/hip_runtime.h>
#include <hip/hip_bf16.h>

typedef __attribute__((ext_vector_type(16))) __bf16 v16bf;
typedef __attribute__((ext_vector_type(8)))  __bf16 v8bf;
typedef __attribute__((ext_vector_type(8)))  float  v8f;
typedef int v4i __attribute__((vector_size(16)));

#define NN    10000
#define DIN   256
#define HH    128
#define DOUT  64
#define NPAD  10016   /* 313 * 32 */
#define GRIDM 625     /* NN / 16  */
#define NSTEP (NPAD / 32)

#if defined(__AMDGCN__) && __has_builtin(__builtin_amdgcn_global_load_async_to_lds_b128)
#define HAVE_ASYNC_LDS 1
#else
#define HAVE_ASYNC_LDS 0
#endif

static __device__ __forceinline__ __bf16 to_bf(float f) { return (__bf16)f; }

#if HAVE_ASYNC_LDS
static __device__ __forceinline__ void async_cp16(const void* g, void* l) {
  __builtin_amdgcn_global_load_async_to_lds_b128(
      (__attribute__((address_space(1))) v4i*)g,
      (__attribute__((address_space(3))) v4i*)l, 0, 0);
}
static __device__ __forceinline__ void wait_async_le2() {
  asm volatile("s_wait_asynccnt 0x2" ::: "memory");
}
static __device__ __forceinline__ void wait_async_0() {
  asm volatile("s_wait_asynccnt 0x0" ::: "memory");
}
#endif

// ---------------------------------------------------------------------------
// degree / norm
// ---------------------------------------------------------------------------
__global__ void zero_f32_kernel(float* __restrict__ p, int n) {
  int i = blockIdx.x * blockDim.x + threadIdx.x;
  if (i < n) p[i] = 0.f;
}

__global__ void deg_partial_kernel(const float* __restrict__ adj, float* __restrict__ deg) {
  int i = blockIdx.x * 256 + threadIdx.x;
  if (i >= NN) return;
  int j0 = blockIdx.y * 200;
  int j1 = j0 + 200; if (j1 > NN) j1 = NN;
  float s = 0.f;
  for (int j = j0; j < j1; ++j) {
    // speculative prefetch 16 rows ahead (global_prefetch_b8); OOB is dropped
    __builtin_prefetch(&adj[(size_t)(j + 16) * NN + i], 0, 0);
    s += adj[(size_t)j * NN + i];   // coalesced over i
  }
  atomicAdd(&deg[i], s);
}

__global__ void norm_finish_kernel(float* __restrict__ norm) {
  int i = blockIdx.x * 256 + threadIdx.x;
  if (i >= NN) return;
  float d = norm[i];
  if (d < 1.f) d = 1.f;
  norm[i] = rsqrtf(d);
}

// ---------------------------------------------------------------------------
// fp32 -> bf16 bulk convert
// ---------------------------------------------------------------------------
__global__ void cvt_bf16_kernel(const float* __restrict__ src, unsigned short* __restrict__ dst, int n) {
  int i = blockIdx.x * blockDim.x + threadIdx.x;
  if (i < n) ((__bf16*)dst)[i] = to_bf(src[i]);
}

// ---------------------------------------------------------------------------
// BnT[f][i] = bf16(norm[i] * X[i][f]), zero-padded for i in [NN, NPAD)
// ---------------------------------------------------------------------------
__global__ void make_bnT_kernel(const float* __restrict__ Xf, const float* __restrict__ norm,
                                unsigned short* __restrict__ BnT) {
  int idx = blockIdx.x * 256 + threadIdx.x;
  if (idx >= HH * NPAD) return;
  int f = idx / NPAD;
  int i = idx - f * NPAD;
  float v = 0.f;
  if (i < NN) v = norm[i] * Xf[(size_t)i * HH + f];
  ((__bf16*)BnT)[idx] = to_bf(v);
}

// ---------------------------------------------------------------------------
// Generic WMMA GEMM:  C[16x16 tile] = act(A_bf16[MxK] @ B_f32[KxNc] + bias)
// 1 wave per block; grid = (M/16, Nc/16).  K multiple of 32.
// A fragment (16-bit A 16x32): lane m=lane&15, g=lane>>4,
//   element e -> K = e + 16g  => one contiguous 32B run (two 16B loads).
// B fragment: lane n=lane&15, element e -> K = e + 16g => one contiguous 32B
//   run served from an LDS-transposed tile of the fp32 weights.
// C/D layout: VGPR r -> row = r + 8g, col = lane&15.
// ---------------------------------------------------------------------------
template <int RELU, int WF32, int WB16>
__global__ __launch_bounds__(32)
void wmma_gemm_kernel(const unsigned short* __restrict__ Au, int lda,
                      const float* __restrict__ B, int ldb,
                      const float* __restrict__ bias, int K,
                      float* __restrict__ Cf, int ldcf,
                      unsigned short* __restrict__ Cbu, int ldcb) {
  __shared__ __bf16 sB[16 * 32] __attribute__((aligned(32)));
  const __bf16* A = (const __bf16*)Au;
  const int lane = threadIdx.x;
  const int g = lane >> 4, m = lane & 15;
  const int i0 = blockIdx.x * 16;
  const int n0 = blockIdx.y * 16;
  v8f c = {};
  for (int k0 = 0; k0 < K; k0 += 32) {
    const float* brow = B + (size_t)(k0 + lane) * ldb + n0;
#pragma unroll
    for (int cc = 0; cc < 16; ++cc) sB[cc * 32 + lane] = to_bf(brow[cc]);
    __syncthreads();
    const __bf16* arow = A + (size_t)(i0 + m) * lda + k0;
    v8bf alo = *(const v8bf*)(arow + g * 16);
    v8bf ahi = *(const v8bf*)(arow + g * 16 + 8);
    v16bf a;
#pragma unroll
    for (int e = 0; e < 8; ++e) { a[e] = alo[e]; a[8 + e] = ahi[e]; }
    v16bf b = *(const v16bf*)(&sB[m * 32 + g * 16]);
    c = __builtin_amdgcn_wmma_f32_16x16x32_bf16(false, a, false, b, (short)0, c, false, false);
    __syncthreads();
  }
#pragma unroll
  for (int r = 0; r < 8; ++r) {
    int row = i0 + r + 8 * g;
    int col = n0 + m;
    float v = c[r] + bias[col];
    if (RELU) v = v > 0.f ? v : 0.f;
    if (WF32) Cf[(size_t)row * ldcf + col] = v;
    if (WB16) ((__bf16*)Cbu)[(size_t)row * ldcb + col] = to_bf(v);
  }
}

// ---------------------------------------------------------------------------
// Propagation GEMM:  P[i][n] = norm[i] * sum_k adj[k][i] * BnT[n][k]
//   out = alpha * P + beta * Xbeta      (X1: alpha=-1; X2: alpha=-2, beta=-1)
// 256 threads = 8 waves; block owns a 16x128 output slab.
//  - A (adj^T) tile: cooperatively staged (coalesced 64B row reads, fp32->bf16
//    VALU convert), double-buffered => adj streamed exactly once per prop.
//  - B (BnT) slab: each wave owns rows n in [16w,16w+16) and stages its own
//    1KB via 2x global_load_async_to_lds_b128 per step, double-buffered and
//    drained with s_wait_asynccnt (no barrier needed for B).
// One barrier per K-step total.
// ---------------------------------------------------------------------------
template <int HASBETA>
__global__ __launch_bounds__(256)
void prop_wmma_kernel(const float* __restrict__ adj,
                      const unsigned short* __restrict__ BnTu,
                      const float* __restrict__ norm,
                      float alpha, float beta,
                      const float* __restrict__ Xbeta,
                      float* __restrict__ Xf,
                      unsigned short* __restrict__ Xcatu, int ldxc) {
  __shared__ __bf16 sA[2][16 * 32] __attribute__((aligned(32)));
#if HAVE_ASYNC_LDS
  __shared__ __bf16 sBn[2][128 * 32] __attribute__((aligned(32)));  // [n][k]
#endif
  const __bf16* BnT = (const __bf16*)BnTu;
  const int t = threadIdx.x;
  const int lane = t & 31;
  const int w = t >> 5;
  const int g = lane >> 4, m = lane & 15;
  const int i0 = blockIdx.x * 16;

  auto stageA = [&](int step, int buf) {
    int k  = t >> 3;          // 0..31
    int c2 = (t & 7) * 2;     // 0,2,...,14
    float2 v = make_float2(0.f, 0.f);
    int kk = step * 32 + k;
    if (kk < NN) v = *(const float2*)(adj + (size_t)kk * NN + i0 + c2);
    sA[buf][c2 * 32 + k]       = to_bf(v.x);
    sA[buf][(c2 + 1) * 32 + k] = to_bf(v.y);
  };
#if HAVE_ASYNC_LDS
  auto stageB = [&](int step, int buf) {
    // wave w stages rows 16w..16w+15; lane covers 2 of the 64 16B chunks
    int r0 = lane >> 2;            // 0..7
    int h  = lane & 3;             // 16B chunk within a 64B row
    int na = w * 16 + r0;
    int nb = w * 16 + 8 + r0;
    const char* gb = (const char*)BnT;
    char* lb = (char*)&sBn[buf][0];
    size_t kb = (size_t)step * 64 + (size_t)h * 16;  // step*32 bf16 = 64 bytes
    async_cp16(gb + (size_t)na * (NPAD * 2) + kb, lb + na * 64 + h * 16);
    async_cp16(gb + (size_t)nb * (NPAD * 2) + kb, lb + nb * 64 + h * 16);
  };
#endif

  stageA(0, 0);
#if HAVE_ASYNC_LDS
  stageB(0, 0);
#endif
  v8f c = {};
  for (int step = 0; step < NSTEP; ++step) {
    const int buf = step & 1, nxt = buf ^ 1;
    __syncthreads();                       // sA[buf] visible to all waves
    const bool more = (step + 1) < NSTEP;
    if (more) {
      stageA(step + 1, nxt);
#if HAVE_ASYNC_LDS
      stageB(step + 1, nxt);
#endif
    }
#if HAVE_ASYNC_LDS
    if (more) wait_async_le2(); else wait_async_0();   // sBn[buf] ready
#endif
    v16bf a;
    {
      v8bf alo = *(const v8bf*)(&sA[buf][m * 32 + g * 16]);
      v8bf ahi = *(const v8bf*)(&sA[buf][m * 32 + g * 16 + 8]);
#pragma unroll
      for (int e = 0; e < 8; ++e) { a[e] = alo[e]; a[8 + e] = ahi[e]; }
    }
#if HAVE_ASYNC_LDS
    v16bf b = *(const v16bf*)(&sBn[buf][(w * 16 + m) * 32 + g * 16]);
#else
    v16bf b = *(const v16bf*)(BnT + (size_t)(w * 16 + m) * NPAD + step * 32 + g * 16);
#endif
    c = __builtin_amdgcn_wmma_f32_16x16x32_bf16(false, a, false, b, (short)0, c, false, false);
  }
#pragma unroll
  for (int r = 0; r < 8; ++r) {
    int row = i0 + r + 8 * g;
    int col = w * 16 + m;
    float p = alpha * norm[row] * c[r];
    if (HASBETA) p += beta * Xbeta[(size_t)row * HH + col];
    Xf[(size_t)row * HH + col] = p;
    ((__bf16*)Xcatu)[(size_t)row * ldxc + col] = to_bf(p);
  }
}

// ---------------------------------------------------------------------------
extern "C" void kernel_launch(void* const* d_in, const int* in_sizes, int n_in,
                              void* d_out, int out_size, void* d_ws, size_t ws_size,
                              hipStream_t stream) {
  (void)in_sizes; (void)n_in; (void)out_size; (void)ws_size;
  const float* x      = (const float*)d_in[0];
  const float* adj    = (const float*)d_in[1];
  const float* W_in   = (const float*)d_in[2];
  const float* b_in   = (const float*)d_in[3];
  const float* W_cheb = (const float*)d_in[4];
  const float* b_cheb = (const float*)d_in[5];
  const float* W_out  = (const float*)d_in[6];
  const float* b_out  = (const float*)d_in[7];
  float* out = (float*)d_out;

  char* p = (char*)d_ws;
  auto alloc = [&](size_t bytes) -> char* {
    char* r = p; p += (bytes + 255) & ~(size_t)255; return r;
  };
  float*          norm = (float*)alloc((size_t)NN * 4);
  unsigned short* xb   = (unsigned short*)alloc((size_t)NN * DIN * 2);
  float*          X0f  = (float*)alloc((size_t)NN * HH * 4);
  float*          X1f  = (float*)alloc((size_t)NN * HH * 4);
  float*          X2f  = (float*)alloc((size_t)NN * HH * 4);
  unsigned short* Xcat = (unsigned short*)alloc((size_t)NN * 3 * HH * 2);
  unsigned short* BnT  = (unsigned short*)alloc((size_t)HH * NPAD * 2);
  unsigned short* hcb  = (unsigned short*)alloc((size_t)NN * HH * 2);

  // 1) degree -> norm
  zero_f32_kernel<<<(NN + 255) / 256, 256, 0, stream>>>(norm, NN);
  deg_partial_kernel<<<dim3(40, 50), 256, 0, stream>>>(adj, norm);
  norm_finish_kernel<<<40, 256, 0, stream>>>(norm);

  // 2) x -> bf16
  cvt_bf16_kernel<<<((NN * DIN) + 255) / 256, 256, 0, stream>>>(x, xb, NN * DIN);

  // 3) X0 = relu(x @ W_in + b_in)  -> X0f (f32) and Xcat[:,0:128] (bf16)
  wmma_gemm_kernel<1, 1, 1><<<dim3(GRIDM, HH / 16), 32, 0, stream>>>(
      xb, DIN, W_in, HH, b_in, DIN, X0f, HH, Xcat + 0, 3 * HH);

  // 4) X1 = -prop(X0)
  make_bnT_kernel<<<((HH * NPAD) + 255) / 256, 256, 0, stream>>>(X0f, norm, BnT);
  prop_wmma_kernel<0><<<GRIDM, 256, 0, stream>>>(
      adj, BnT, norm, -1.f, 0.f, nullptr, X1f, Xcat + HH, 3 * HH);

  // 5) X2 = -2*prop(X1) - X0
  make_bnT_kernel<<<((HH * NPAD) + 255) / 256, 256, 0, stream>>>(X1f, norm, BnT);
  prop_wmma_kernel<1><<<GRIDM, 256, 0, stream>>>(
      adj, BnT, norm, -2.f, -1.f, X0f, X2f, Xcat + 2 * HH, 3 * HH);

  // 6) hc = relu([X0 X1 X2] @ W_cheb + b_cheb)  (bf16 only)
  wmma_gemm_kernel<1, 0, 1><<<dim3(GRIDM, HH / 16), 32, 0, stream>>>(
      Xcat, 3 * HH, W_cheb, HH, b_cheb, 3 * HH, nullptr, 0, hcb, HH);

  // 7) out = hc @ W_out + b_out  (f32 to d_out)
  wmma_gemm_kernel<0, 1, 0><<<dim3(GRIDM, DOUT / 16), 32, 0, stream>>>(
      hcb, HH, W_out, DOUT, b_out, HH, out, DOUT, nullptr, 0);
}
